// Decoder_24043226923528
// MI455X (gfx1250) — matrix operand order repaired and checked by reading, not compiled, not run on hardware
//
#include <hip/hip_runtime.h>
#include <hip/hip_bf16.h>
#include <math.h>

// ---------------------------------------------------------------------------
// 2-layer tanh RNN (B=64, T=256, D=H=512) on gfx1250 via bf16 WMMA.
//
//   c @ W = x_t @ W_x + h_{t-1} @ W_h
//   - x-part, all timesteps: parallel GEMM [16384,512]x[512,1024], 4 N-tiles
//     per wave, output in WMMA C-fragment layout (b128 per lane).
//   - recurrence: ONE persistent kernel per layer (32 blocks x 8 waves,
//     one 16x16 tile per wave), 256-step time loop inside, device-wide
//     atomic barrier between steps, W_rec slice staged in LDS per block.
// ---------------------------------------------------------------------------

typedef __bf16 bf16_t;
typedef __attribute__((ext_vector_type(16))) __bf16 v16bf;
typedef __attribute__((ext_vector_type(8)))  __bf16 v8bf;
typedef __attribute__((ext_vector_type(8)))  float  v8f;
typedef __attribute__((ext_vector_type(4)))  unsigned int v4u;

#define TT 256   // timesteps
#define BB 64    // batch
#define KK 512   // reduction dim (D == H == 512)
#define NN 1024  // fused output cols (512 h-part + 512 o-part)
#define SCAN_BLOCKS 32

// ---- WMMA fragment helpers -------------------------------------------------

// A fragment (16x32 bf16): lane = m + 16*hi; per lane two contiguous 16-byte
// chunks at K = 16*hi and K = 16*hi + 8 (contiguous K = 16*hi + s, matches B).
__device__ inline v16bf load_a_bf16(const bf16_t* row, int hi) {
    union { v16bf bf; v4u q[2]; } a;
    a.q[0] = *(const v4u*)(row + 16 * hi);
    a.q[1] = *(const v4u*)(row + 16 * hi + 8);
    return a.bf;
}

// Same fragment, converting from f32 source on the fly (lowers to
// global_load_b128 x4 + v_cvt_pk_bf16_f32).
__device__ inline v16bf load_a_f32(const float* row, int hi) {
    v16bf a;
    const float* p0 = row + 16 * hi;
    const float* p1 = row + 16 * hi + 8;
#pragma unroll
    for (int i = 0; i < 8; ++i) {
        a[i]     = (bf16_t)p0[i];
        a[8 + i] = (bf16_t)p1[i];
    }
    return a;
}

// B fragment (32x16 bf16) from transposed weights Wt[N][K]:
// lanes 0-15: n = n0+lane, K 0..15; lanes 16-31: n = n0+lane-16, K 16..31.
__device__ inline v16bf load_b(const bf16_t* wt, int n0, int kbase, int lane) {
    const bf16_t* p = wt + (size_t)(n0 + (lane & 15)) * KK + kbase + ((lane >> 4) << 4);
    return *(const v16bf*)p;
}

__device__ inline v8f wmma_bf16(v16bf a, v16bf b, v8f c) {
    return __builtin_amdgcn_wmma_f32_16x16x32_bf16(
        false, a, false, b, (short)0, c, false, false);
}

// ---- Kernel 1: transpose + convert weights to bf16 [N][K] ------------------
__global__ void prep_weights(const float* __restrict__ Wh0, const float* __restrict__ Wo0,
                             const float* __restrict__ Wh1, const float* __restrict__ Wo1,
                             bf16_t* __restrict__ Wx0t, bf16_t* __restrict__ Wh0t,
                             bf16_t* __restrict__ Wx1t, bf16_t* __restrict__ Wh1t) {
    int idx = blockIdx.x * 256 + threadIdx.x;          // 4 * 1024 * 512 total
    int q   = idx >> 19;
    int rem = idx & ((1 << 19) - 1);
    int n = rem >> 9;                                  // out col 0..1023
    int k = rem & 511;                                 // K 0..511
    int srcrow = ((q & 1) == 0) ? k : (512 + k);       // x-rows vs h-rows
    const float* Wh = (q < 2) ? Wh0 : Wh1;
    const float* Wo = (q < 2) ? Wo0 : Wo1;
    float v = (n < 512) ? Wh[(size_t)srcrow * 512 + n]
                        : Wo[(size_t)srcrow * 512 + (n - 512)];
    bf16_t* dst = (q == 0) ? Wx0t : (q == 1) ? Wh0t : (q == 2) ? Wx1t : Wh1t;
    dst[(size_t)n * KK + k] = (bf16_t)v;
}

// ---- Kernel 2: init hidden state + zero the scan barrier counter -----------
__global__ void init_state(const float* __restrict__ enc, bf16_t* __restrict__ h,
                           int* __restrict__ sync, int mode) {
    int i = blockIdx.x * 256 + threadIdx.x;            // 64*512
    h[i] = (mode == 0) ? (bf16_t)enc[i] : (bf16_t)0.0f;
    if (i == 0) *sync = 0;
}

// ---- Kernel 3: parallel x-part GEMM, 4 N-tiles per wave --------------------
// Pf fragment-major: tile_id = mt*64 + nt, element = lane*8 + j.
// MODE 0: A row (t,b) = x[b][t][:] f32.  MODE 1: A row r = out0[r][:] bf16.
template <int MODE>
__global__ void gemm_precomp(const float* __restrict__ x, const bf16_t* __restrict__ a_bf,
                             const bf16_t* __restrict__ wt, bf16_t* __restrict__ Pf) {
    int wid  = (blockIdx.x * blockDim.x + threadIdx.x) >> 5;   // 0..16383
    int lane = threadIdx.x & 31;
    int mt = wid >> 4;                 // 0..1023  (M tile)
    int n0 = (wid & 15) << 6;          // N group base: 4 tiles of 16
    int hi = lane >> 4;
    int m  = (mt << 4) + (lane & 15);  // global row = t*64 + b

    const float*  arow_f = nullptr;
    const bf16_t* arow_b = nullptr;
    if (MODE == 0) {
        int t = m >> 6, b = m & 63;
        arow_f = x + ((size_t)b * TT + t) * KK;
    } else {
        arow_b = a_bf + (size_t)m * KK;
    }

    v8f acc0 = {}, acc1 = {}, acc2 = {}, acc3 = {};
#pragma unroll
    for (int kt = 0; kt < KK / 32; ++kt) {
        v16bf a = (MODE == 0) ? load_a_f32(arow_f + kt * 32, hi)
                              : load_a_bf16(arow_b + kt * 32, hi);
        acc0 = wmma_bf16(a, load_b(wt, n0,      kt * 32, lane), acc0);
        acc1 = wmma_bf16(a, load_b(wt, n0 + 16, kt * 32, lane), acc1);
        acc2 = wmma_bf16(a, load_b(wt, n0 + 32, kt * 32, lane), acc2);
        acc3 = wmma_bf16(a, load_b(wt, n0 + 48, kt * 32, lane), acc3);
    }

    size_t base = ((size_t)mt * 64 + (n0 >> 4)) * 256 + lane * 8;
    v8f accs[4] = {acc0, acc1, acc2, acc3};
#pragma unroll
    for (int i = 0; i < 4; ++i) {
        v8bf o;
#pragma unroll
        for (int j = 0; j < 8; ++j) o[j] = (bf16_t)accs[i][j];
        *(v8bf*)(Pf + base + (size_t)i * 256) = o;     // one b128 store
    }
}

// ---- Kernel 4: persistent recurrent scan (one launch per layer) ------------
// Per step: out[b][n] = tanh(Pf(tile) + bias[n] + sum_k h[b][k] * Wrec[n][k]).
// 256 waves = 32 blocks x 8; block b owns n-tiles {2b, 2b+1} for ALL steps,
// so its 32 W_rec rows (32 KB) are staged in LDS once. Device-wide barrier
// (monotonic counter, release/acquire) separates steps; h is double-buffered.
template <int LAYER>
__global__ void rnn_scan(const bf16_t* __restrict__ wrec, const bf16_t* __restrict__ Pf,
                         const float* __restrict__ bh, const float* __restrict__ bo,
                         bf16_t* __restrict__ hA, bf16_t* __restrict__ hB,
                         bf16_t* __restrict__ out0, float* __restrict__ dout,
                         int* __restrict__ sync) {
    __shared__ bf16_t wlds[32 * KK];                   // 32 KB: this block's W rows

    int tid  = threadIdx.x;
    int lane = tid & 31;
    int wid  = blockIdx.x * 8 + (tid >> 5);            // 0..255
    int mt   = wid & 3;                                // 4 M tiles (B=64)
    int nt   = wid >> 2;                               // 64 N tiles
    int n0   = nt << 4;
    int hi   = lane >> 4;
    int nblk = blockIdx.x * 32;                        // first W row staged here

    // Stage 32 rows of Wrec into LDS (each thread: 8 x 16B copies).
    {
        const bf16_t* src = wrec + (size_t)nblk * KK;
        for (int i = tid; i < 32 * KK / 8; i += 256)
            *(v4u*)(wlds + i * 8) = *(const v4u*)(src + i * 8);
    }
    __syncthreads();

    // Per-wave step-invariants.
    const bf16_t* wrow = wlds + (size_t)((n0 - nblk) + (lane & 15)) * KK + (hi << 4);
    int   n     = n0 + (lane & 15);
    int   mbase = (mt << 4) + (hi << 3);
    float bias  = (n < 512) ? bh[n] : bo[n - 512];
    int   arow_off = ((mt << 4) + (lane & 15)) * KK;
    const bf16_t* ptile = Pf + ((size_t)mt * 64 + nt) * 256 + lane * 8;

#pragma unroll 1
    for (int t = 0; t < TT; ++t) {
        const bf16_t* h_in  = (t & 1) ? hB : hA;
        bf16_t*       h_out = (t & 1) ? hA : hB;
        const bf16_t* arow  = h_in + arow_off;

        v8f acc = {};
#pragma unroll
        for (int kt = 0; kt < KK / 32; ++kt) {
            v16bf a = load_a_bf16(arow + kt * 32, hi);
            union { v16bf bf; v4u q[2]; } b;           // B frag from LDS
            b.q[0] = *(const v4u*)(wrow + kt * 32);
            b.q[1] = *(const v4u*)(wrow + kt * 32 + 8);
            acc = wmma_bf16(a, b.bf, acc);
        }

        v8bf p8 = *(const v8bf*)(ptile + (size_t)t * 4 * 64 * 256);
#pragma unroll
        for (int j = 0; j < 8; ++j) {
            int m = mbase + j;                         // batch index 0..63
            float v  = acc[j] + (float)p8[j] + bias;
            float tv = tanhf(v);
            if (n < 512) {
                h_out[(size_t)m * KK + n] = (bf16_t)tv;
            } else if (LAYER == 0) {
                out0[((size_t)t * BB + m) * KK + (n - 512)] = (bf16_t)tv;
            } else {
                dout[((size_t)m * TT + t) * KK + (n - 512)] = tv;  // [B,T,D]
            }
        }

        // ---- device-wide step barrier (monotonic counter) ----
        __syncthreads();
        __threadfence();                               // release this step's writes
        if (tid == 0) {
            __hip_atomic_fetch_add(sync, 1, __ATOMIC_RELEASE, __HIP_MEMORY_SCOPE_AGENT);
            int target = SCAN_BLOCKS * (t + 1);
            while (__hip_atomic_load(sync, __ATOMIC_ACQUIRE, __HIP_MEMORY_SCOPE_AGENT) < target)
                __builtin_amdgcn_s_sleep(1);
        }
        __syncthreads();
        __threadfence();                               // acquire: drop stale h lines
    }
}

// ---- Host-side launch chain -------------------------------------------------
extern "C" void kernel_launch(void* const* d_in, const int* in_sizes, int n_in,
                              void* d_out, int out_size, void* d_ws, size_t ws_size,
                              hipStream_t stream) {
    const float* x   = (const float*)d_in[0];
    const float* enc = (const float*)d_in[1];
    const float* Wh0 = (const float*)d_in[2];
    const float* bh0 = (const float*)d_in[3];
    const float* Wo0 = (const float*)d_in[4];
    const float* bo0 = (const float*)d_in[5];
    const float* Wh1 = (const float*)d_in[6];
    const float* bh1 = (const float*)d_in[7];
    const float* Wo1 = (const float*)d_in[8];
    const float* bo1 = (const float*)d_in[9];
    float* out = (float*)d_out;
    char*  ws  = (char*)d_ws;

    const size_t MB = 1024 * 1024;
    bf16_t* Wx0t = (bf16_t*)(ws + 0 * MB);       // 1 MB each [1024][512] bf16
    bf16_t* Wh0t = (bf16_t*)(ws + 1 * MB);
    bf16_t* Wx1t = (bf16_t*)(ws + 2 * MB);
    bf16_t* Wh1t = (bf16_t*)(ws + 3 * MB);
    bf16_t* hA   = (bf16_t*)(ws + 4 * MB);       // 64 KB
    bf16_t* hB   = (bf16_t*)(ws + 4 * MB + 64 * 1024);
    int*    sync = (int*)   (ws + 4 * MB + 128 * 1024);
    bf16_t* out0 = (bf16_t*)(ws + 5 * MB);       // 16 MB  [16384][512]
    bf16_t* Pf   = (bf16_t*)(ws + 24 * MB);      // 32 MB  fragment-major
    // total workspace footprint: 56 MB

    prep_weights<<<8192, 256, 0, stream>>>(Wh0, Wo0, Wh1, Wo1, Wx0t, Wh0t, Wx1t, Wh1t);

    // ---- layer 0 ----
    init_state<<<128, 256, 0, stream>>>(enc, hA, sync, 0);
    gemm_precomp<0><<<2048, 256, 0, stream>>>(x, nullptr, Wx0t, Pf);
    rnn_scan<0><<<SCAN_BLOCKS, 256, 0, stream>>>(Wh0t, Pf, bh0, bo0, hA, hB, out0, nullptr, sync);

    // ---- layer 1 ----
    init_state<<<128, 256, 0, stream>>>(nullptr, hA, sync, 1);
    gemm_precomp<1><<<2048, 256, 0, stream>>>(nullptr, out0, Wx1t, Pf);
    rnn_scan<1><<<SCAN_BLOCKS, 256, 0, stream>>>(Wh1t, Pf, bh1, bo1, hA, hB, nullptr, out, sync);

    (void)in_sizes; (void)n_in; (void)out_size; (void)ws_size;
}
